// CondConv2d_39633958207548
// MI455X (gfx1250) — matrix operand order repaired and checked by reading, not compiled, not run on hardware
//
#include <hip/hip_runtime.h>

// ---------------------------------------------------------------------------
// CondConv2d for MI455X (gfx1250, wave32, WMMA).
// Identity: weight scaled on input channels == input scaled per channel.
//   out[n] = conv2d(x[n] * scale[n,ci], weight) + bias * (gate[n]+1)
// Main conv = implicit GEMM with v_wmma_f32_16x16x32_f16 (f32 accum).
// Staging uses CDNA5 async global->LDS DMA (global_load_async_to_lds_b128,
// ASYNCcnt) into double-buffered LDS tiles; input pre-scaled to f16 with an
// explicit zero halo so tap-shifted loads are branch-free. Chunk loop is
// rolled (unroll disabled) + last chunk peeled so accumulators stay pinned.
// ---------------------------------------------------------------------------

typedef __attribute__((ext_vector_type(16))) _Float16 v16h;
typedef __attribute__((ext_vector_type(8)))  _Float16 v8h;
typedef __attribute__((ext_vector_type(8)))  float    v8f;

#define NB    16
#define NL    4
#define NSAMP 64         // NB*NL
#define CIN   256
#define COUT  256
#define HW    1024       // 32*32
#define XH    34         // 32 + halo
#define LDSW  40         // LDS row stride in halves (32 data + 8 pad)
#define NCHUNK 72        // 9 taps * 8 ci-blocks of 32

#define USE_ASYNC 1      // CDNA5 async global->LDS staging (inline asm)

// ---- K1: pooled[n][c] = mean_{h,w} x[n][c][h][w] ; one wave per (n,c) ----
__global__ __launch_bounds__(256) void cc_pool(const float* __restrict__ x,
                                               float* __restrict__ pooled) {
    int wg   = blockIdx.x * 8 + (threadIdx.x >> 5);   // wave id in [0, 64*256)
    int lane = threadIdx.x & 31;
    const float* src = x + (size_t)wg * HW;
    float s = 0.f;
    #pragma unroll 4
    for (int i = 0; i < 32; ++i) s += src[lane + 32 * i];
    #pragma unroll
    for (int off = 16; off > 0; off >>= 1) s += __shfl_xor(s, off, 32);
    if (lane == 0) pooled[wg] = s * (1.0f / 1024.0f);
}

// ---- K2: scale[n][c] = calib[b,c,l] + 1 (temporal conv, first frame dup) ----
__global__ __launch_bounds__(256) void cc_calib(const float* __restrict__ pooled,
                                                const float* __restrict__ tw,
                                                const float* __restrict__ tb,
                                                float* __restrict__ scale) {
    int idx = blockIdx.x * blockDim.x + threadIdx.x;  // b*256*4
    if (idx >= NB * CIN * NL) return;
    int l = idx & 3;
    int c = (idx >> 2) & (CIN - 1);
    int b = idx >> 10;
    float acc = tb[c];
    #pragma unroll
    for (int k = 0; k < 3; ++k) {
        int s = l + k - 2; if (s < 0) s = 0;              // duplicated frame 0
        const float* p = pooled + ((b * NL + s) * CIN);
        const float* w = tw + (size_t)c * CIN * 3 + k;
        for (int ci = 0; ci < CIN; ++ci) acc += p[ci] * w[ci * 3];
    }
    scale[(b * NL + l) * CIN + c] = acc + 1.0f;
}

// ---- K3: gp1[n] = gate[b,l] + 1 ----
__global__ __launch_bounds__(64) void cc_gate(const float* __restrict__ pooled,
                                              const float* __restrict__ fw,
                                              const float* __restrict__ fb,
                                              float* __restrict__ gp1) {
    int n = blockIdx.x * blockDim.x + threadIdx.x;
    if (n >= NSAMP) return;
    int b = n >> 2, l = n & 3;
    float acc = fb[0];
    #pragma unroll
    for (int k = 0; k < 3; ++k) {
        int s = l + k - 2; if (s < 0) s = 0;
        const float* p = pooled + ((b * NL + s) * CIN);
        for (int ci = 0; ci < 256; ++ci) acc += p[ci] * fw[ci * 3 + k];
    }
    gp1[n] = acc + 1.0f;
}

// ---- K4: xs[n][yy][xx][ci] = f16(x * scale) with 1-pixel zero halo ----
// grid (34*34, NSAMP), 256 threads = one ci-row per block (coalesced writes)
__global__ __launch_bounds__(256) void cc_xscale(const float* __restrict__ x,
                                                 const float* __restrict__ scale,
                                                 _Float16* __restrict__ xs) {
    const int yy = blockIdx.x / XH;
    const int xx = blockIdx.x - yy * XH;
    const int n  = blockIdx.y;
    const int ci = threadIdx.x;
    const int sy = yy - 1, sx = xx - 1;
    float v = 0.f;
    if ((unsigned)sy < 32u && (unsigned)sx < 32u)
        v = x[((size_t)(n * CIN + ci) << 10) + sy * 32 + sx] * scale[n * CIN + ci];
    xs[(((size_t)n * XH + yy) * XH + xx) * CIN + ci] = (_Float16)v;
}

// ---- K5: w2[r][co][ci] = f16(weight[co][ci][kh][kw]), r = kh*3+kw ----
__global__ __launch_bounds__(256) void cc_wcvt(const float* __restrict__ w,
                                               _Float16* __restrict__ w2) {
    int idx = blockIdx.x * blockDim.x + threadIdx.x;  // 9*256*256
    if (idx >= 9 * COUT * CIN) return;
    int ci = idx & 255;
    int co = (idx >> 8) & 255;
    int r  = idx >> 16;
    w2[idx] = (_Float16)w[(size_t)(co * CIN + ci) * 9 + r];
}

// ---- CDNA5 async global->LDS helpers ----
__device__ __forceinline__ void async_ld16(unsigned lds_addr, const void* gaddr) {
#if USE_ASYNC
    asm volatile("global_load_async_to_lds_b128 %0, %1, off"
                 :: "v"(lds_addr), "v"(gaddr) : "memory");
#endif
}
__device__ __forceinline__ void async_wait0() {
#if __has_builtin(__builtin_amdgcn_s_wait_asynccnt)
    __builtin_amdgcn_s_wait_asynccnt(0);
#else
    asm volatile("s_wait_asynccnt 0x0" ::: "memory");
#endif
}
__device__ __forceinline__ v16h cat16(v8h lo, v8h hi) {
    return __builtin_shufflevector(lo, hi, 0, 1, 2, 3, 4, 5, 6, 7,
                                           8, 9, 10, 11, 12, 13, 14, 15);
}

// ---- Main: implicit GEMM, per sample: out[co,p] = sum_r,ci w2 * xs(shift r) ----
// grid (8 pixel-tiles, 2 co-tiles, 64 samples); 256 threads = 8 waves.
// Block tile 128(co) x 128(pix); wave tile 32(co) x 64(pix) = 2x4 WMMA frags.
__global__ __launch_bounds__(256) void cc_gemm(const _Float16* __restrict__ xs,
                                               const _Float16* __restrict__ w2,
                                               const float* __restrict__ bias,
                                               const float* __restrict__ gp1,
                                               float* __restrict__ out) {
    __shared__ __align__(16) _Float16 Ash[2][128 * LDSW];  // [buf][co_local][k]
    __shared__ __align__(16) _Float16 Bsh[2][128 * LDSW];  // [buf][pix_local][k]

    const int tid  = threadIdx.x;
    const int lane = tid & 31;
    const int wave = tid >> 5;
    const int wco  = wave & 3;   // 4 waves along co  (32 each)
    const int wp   = wave >> 2;  // 2 waves along pix (64 each)

    const int ptile = blockIdx.x;        // pixel tile: 4 image rows
    const int co0   = blockIdx.y * 128;  // co tile
    const int n     = blockIdx.z;        // sample
    const int y0    = ptile * 4;

    // per-thread staging assignment: rows rowA, rowA+64; 16B segment kseg
    const int rowA = tid >> 2;           // 0..63
    const int kseg = (tid & 3) * 8;      // half offset within 32-wide K chunk

    // thread-invariant global bases (per-chunk delta is a uniform scalar add)
    const _Float16* gA0 = w2 + (size_t)(co0 + rowA) * CIN + kseg;
    const _Float16* gA1 = gA0 + (size_t)64 * CIN;
    const int rB1 = rowA + 64;
    const _Float16* gB0 = xs + (((size_t)n * XH + y0 + (rowA >> 5)) * XH + (rowA & 31)) * CIN + kseg;
    const _Float16* gB1 = xs + (((size_t)n * XH + y0 + (rB1 >> 5)) * XH + (rB1 & 31)) * CIN + kseg;

    // thread-invariant LDS destinations (both buffers)
    unsigned lA[2][2], lB[2][2];
    #pragma unroll
    for (int buf = 0; buf < 2; ++buf)
        #pragma unroll
        for (int it = 0; it < 2; ++it) {
            lA[buf][it] = (unsigned)(size_t)&Ash[buf][(rowA + it * 64) * LDSW + kseg];
            lB[buf][it] = (unsigned)(size_t)&Bsh[buf][(rowA + it * 64) * LDSW + kseg];
        }

    // stage one chunk (r = tap, cib = ci block) into LDS buffer `buf`
    auto stage = [&](int chunk, int buf) {
        const int r   = chunk >> 3;
        const int cib = chunk & 7;
        const int kh  = r / 3;                                        // uniform
        const int kw  = r - 3 * kh;
        const size_t offA = (size_t)r * COUT * CIN + cib * 32;        // uniform
        const size_t offB = (size_t)(kh * XH + kw) * CIN + cib * 32;  // uniform
#if USE_ASYNC
        async_ld16(lA[buf][0], gA0 + offA);
        async_ld16(lA[buf][1], gA1 + offA);
        async_ld16(lB[buf][0], gB0 + offB);
        async_ld16(lB[buf][1], gB1 + offB);
#else
        v8h a0 = *(const v8h*)(gA0 + offA);
        v8h a1 = *(const v8h*)(gA1 + offA);
        v8h b0 = *(const v8h*)(gB0 + offB);
        v8h b1 = *(const v8h*)(gB1 + offB);
        *(v8h*)(size_t)lA[buf][0] = a0;
        *(v8h*)(size_t)lA[buf][1] = a1;
        *(v8h*)(size_t)lB[buf][0] = b0;
        *(v8h*)(size_t)lB[buf][1] = b1;
#endif
    };

    v8f acc[2][4];
    #pragma unroll
    for (int a = 0; a < 2; ++a)
        #pragma unroll
        for (int b = 0; b < 4; ++b) acc[a][b] = v8f{};

    const int mrow = lane & 15;
    const int hi   = lane >> 4;

    // fragment compute from buffer `cur` (inlined twice: loop body + peel)
    auto compute = [&](int cur) {
        // Fragments per ISA VGPR layouts (CDNA5 ISA):
        // A 16x32: lanes 0-15 K=0..7,16..23 ; lanes 16-31 K=8..15,24..31
        // B 32x16: lanes 0-15 K=0..7,16..23 ; lanes 16-31 K=8..15,24..31 (col = lane&15)
        v16h af[2], bf[4];
        #pragma unroll
        for (int a = 0; a < 2; ++a) {
            const int row = wco * 32 + a * 16 + mrow;
            const int k0  = hi * 8;
            af[a] = cat16(*(const v8h*)(&Ash[cur][row * LDSW + k0]),
                          *(const v8h*)(&Ash[cur][row * LDSW + k0 + 16]));
        }
        #pragma unroll
        for (int b = 0; b < 4; ++b) {
            const int row = wp * 64 + b * 16 + mrow;
            const int k0  = hi * 8;
            bf[b] = cat16(*(const v8h*)(&Bsh[cur][row * LDSW + k0]),
                          *(const v8h*)(&Bsh[cur][row * LDSW + k0 + 16]));
        }
        #pragma unroll
        for (int a = 0; a < 2; ++a)
            #pragma unroll
            for (int b = 0; b < 4; ++b)
                acc[a][b] = __builtin_amdgcn_wmma_f32_16x16x32_f16(
                    false, af[a], false, bf[b],
                    (short)0, acc[a][b], false, false);
    };

    // prologue: stage chunk 0 into buffer 0
    stage(0, 0);
    async_wait0();
    __syncthreads();

    // rolled steady-state loop: unconditional body, accumulators stay pinned
    #pragma clang loop unroll(disable)
    for (int idx = 0; idx < NCHUNK - 1; ++idx) {
        const int cur = idx & 1;
        stage(idx + 1, cur ^ 1);   // DMA for next chunk runs under WMMAs
        compute(cur);
        async_wait0();             // our DMA into next buffer is done
        __syncthreads();           // everyone's DMA visible
    }
    // peeled final chunk (no staging, no barrier)
    compute((NCHUNK - 1) & 1);

    // Epilogue: out[n][co][p] = acc + bias[co] * gp1[n]
    const float g = gp1[n];
    #pragma unroll
    for (int a = 0; a < 2; ++a) {
        #pragma unroll
        for (int b = 0; b < 4; ++b) {
            #pragma unroll
            for (int i = 0; i < 8; ++i) {
                int co = co0 + wco * 32 + a * 16 + i + 8 * hi;
                int p  = ptile * 128 + wp * 64 + b * 16 + mrow;
                out[((size_t)n * COUT + co) * HW + p] = acc[a][b][i] + bias[co] * g;
            }
        }
    }
}

// ---------------------------------------------------------------------------
extern "C" void kernel_launch(void* const* d_in, const int* in_sizes, int n_in,
                              void* d_out, int out_size, void* d_ws, size_t ws_size,
                              hipStream_t stream) {
    const float* x       = (const float*)d_in[0];  // (16,4,256,32,32)
    const float* weight  = (const float*)d_in[1];  // (256,256,3,3)
    const float* bias    = (const float*)d_in[2];  // (256,)
    const float* tconv_w = (const float*)d_in[3];  // (256,256,3)
    const float* tconv_b = (const float*)d_in[4];  // (256,)
    const float* fc_w    = (const float*)d_in[5];  // (1,256,3)
    const float* fc_b    = (const float*)d_in[6];  // (1,)
    float* out = (float*)d_out;                    // (64,256,32,32)

    // Workspace layout
    const size_t xs_bytes = (size_t)NSAMP * XH * XH * CIN * 2;  // ~36.1 MB (halo)
    const size_t w2_bytes = (size_t)9 * COUT * CIN * 2;         // 1.18 MB
    char* ws = (char*)d_ws;
    _Float16* xs = (_Float16*)ws;
    _Float16* w2 = (_Float16*)(ws + xs_bytes);
    float* pooled = (float*)(ws + xs_bytes + w2_bytes);
    float* scale  = pooled + NSAMP * CIN;
    float* gp1    = scale + NSAMP * CIN;

    // Prologue (all tiny)
    cc_pool  <<<dim3(NSAMP * CIN / 8), dim3(256), 0, stream>>>(x, pooled);
    cc_calib <<<dim3(NB * CIN * NL / 256), dim3(256), 0, stream>>>(pooled, tconv_w, tconv_b, scale);
    cc_gate  <<<dim3(1), dim3(64), 0, stream>>>(pooled, fc_w, fc_b, gp1);
    cc_xscale<<<dim3(XH * XH, NSAMP), dim3(256), 0, stream>>>(x, scale, xs);
    cc_wcvt  <<<dim3(9 * COUT * CIN / 256), dim3(256), 0, stream>>>(weight, w2);

    // Main implicit-GEMM conv: 8 pixel tiles x 2 co tiles x 64 samples
    cc_gemm  <<<dim3(8, 2, NSAMP), dim3(256), 0, stream>>>(xs, w2, bias, gp1, out);
}
